// DotProductAttention_50268297232509
// MI455X (gfx1250) — matrix operand order repaired and checked by reading, not compiled, not run on hardware
//
#include <hip/hip_runtime.h>
#include <hip/hip_bf16.h>

typedef _Float16 v16h __attribute__((ext_vector_type(16)));
typedef _Float16 v8h  __attribute__((ext_vector_type(8)));
typedef float    v8f  __attribute__((ext_vector_type(8)));

#define NEGV   (-1000000.0f)
#define SCALE  (0.08838834764831845f)   // 1/sqrt(128)

// load halfs p[0..7] and p[16..23] (16B-aligned) from LDS as two b128 loads
__device__ __forceinline__ v16h load16h(const _Float16* p) {
    v8h lo = *(const v8h*)p;
    v8h hi = *(const v8h*)(p + 16);
    v16h r;
#pragma unroll
    for (int i = 0; i < 8; ++i) { r[i] = lo[i]; r[i + 8] = hi[i]; }
    return r;
}

__global__ __launch_bounds__(256, 1)
void fa_f16_wmma_kernel(const float* __restrict__ Qp,
                        const float* __restrict__ Kp,
                        const float* __restrict__ Vp,
                        const int*   __restrict__ Mp,
                        float* __restrict__ Op)
{
    // ---- LDS (double-buffered K/V tiles) ----
    __shared__ __align__(32) _Float16 sK[2][32 * 136];   // K tile, row-major [key][d]
    __shared__ __align__(32) _Float16 sVt[2][128 * 40];  // V tile transposed [d][key]
    __shared__ __align__(16) float    sS[8][16 * 33];    // per-wave S scratch
    __shared__ __align__(32) _Float16 sP[8][16 * 40];    // per-wave P (fp16, row-major)
    __shared__ float sAlpha[8][16];
    __shared__ float sL[8][16];
    __shared__ float sMaskF[2][32];

    const int tid  = threadIdx.x;
    const int w    = tid >> 5;       // wave in block
    const int lane = tid & 31;
    const int hb   = lane >> 4;      // half-wave (0/1)
    const int ln   = lane & 15;
    const int b    = blockIdx.y;     // batch
    const int qbase = blockIdx.x * 128 + w * 16;

    // cooperative-load mapping: each thread owns 16 consecutive f32 of the tile
    const int trow = (tid * 16) >> 7;        // key within tile (0..31)
    const int tcol = (tid * 16) & 127;       // d offset (multiple of 16)
    const float* kTileBase = Kp + ((size_t)b * 2048 + trow) * 128 + tcol;
    const float* vTileBase = Vp + ((size_t)b * 2048 + trow) * 128 + tcol;

    // ---- preload Q tile as fp16 A-operands (4 chunks of K=32 along D) ----
    v16h aQ[4];
    {
        const float* qrow = Qp + ((size_t)b * 2048 + qbase + ln) * 128;
#pragma unroll
        for (int dc = 0; dc < 4; ++dc) {
#pragma unroll
            for (int g = 0; g < 2; ++g) {
                const int d0 = dc * 32 + g * 16 + hb * 8;
                float4 f0 = *(const float4*)(qrow + d0);
                float4 f1 = *(const float4*)(qrow + d0 + 4);
                aQ[dc][g * 8 + 0] = (_Float16)f0.x;
                aQ[dc][g * 8 + 1] = (_Float16)f0.y;
                aQ[dc][g * 8 + 2] = (_Float16)f0.z;
                aQ[dc][g * 8 + 3] = (_Float16)f0.w;
                aQ[dc][g * 8 + 4] = (_Float16)f1.x;
                aQ[dc][g * 8 + 5] = (_Float16)f1.y;
                aQ[dc][g * 8 + 6] = (_Float16)f1.z;
                aQ[dc][g * 8 + 7] = (_Float16)f1.w;
            }
        }
    }

    v8f oc[8];
#pragma unroll
    for (int dt = 0; dt < 8; ++dt)
#pragma unroll
        for (int r = 0; r < 8; ++r) oc[dt][r] = 0.0f;

    float m_run = -3.0e38f;   // running row max (valid in lanes 0..15, row=lane)
    float l_run = 0.0f;       // running row sum

    // prefetch registers (raw f32 — converted only at commit time)
    float4 kf[4], vf[4];
    int mraw = 1;

    // ---- prologue: fetch + commit tile 0, publish ----
    {
#pragma unroll
        for (int g = 0; g < 4; ++g) {
            kf[g] = *(const float4*)(kTileBase + g * 4);
            vf[g] = *(const float4*)(vTileBase + g * 4);
        }
        if (tid < 32) mraw = Mp[(size_t)b * 2048 + tid];

        _Float16 th[16];
#pragma unroll
        for (int g = 0; g < 4; ++g) {
            th[g * 4 + 0] = (_Float16)kf[g].x; th[g * 4 + 1] = (_Float16)kf[g].y;
            th[g * 4 + 2] = (_Float16)kf[g].z; th[g * 4 + 3] = (_Float16)kf[g].w;
        }
        *(v8h*)&sK[0][trow * 136 + tcol]     = *(v8h*)&th[0];
        *(v8h*)&sK[0][trow * 136 + tcol + 8] = *(v8h*)&th[8];
#pragma unroll
        for (int g = 0; g < 4; ++g) {
            sVt[0][(tcol + g * 4 + 0) * 40 + trow] = (_Float16)vf[g].x;
            sVt[0][(tcol + g * 4 + 1) * 40 + trow] = (_Float16)vf[g].y;
            sVt[0][(tcol + g * 4 + 2) * 40 + trow] = (_Float16)vf[g].z;
            sVt[0][(tcol + g * 4 + 3) * 40 + trow] = (_Float16)vf[g].w;
        }
        if (tid < 32) sMaskF[0][tid] = (mraw != 0) ? 1.0f : 0.0f;
    }
    __syncthreads();

    for (int it = 0; it < 64; ++it) {
        const int cur = it & 1;
        const int nxt = cur ^ 1;
        const bool more = (it < 63);

        // ---- prefetch NEXT tile into registers (consumed only at commit) ----
        if (more) {
            const size_t koff = (size_t)(it + 1) * 32 * 128;
#pragma unroll
            for (int g = 0; g < 4; ++g) {
                kf[g] = *(const float4*)(kTileBase + koff + g * 4);
                vf[g] = *(const float4*)(vTileBase + koff + g * 4);
            }
            if (tid < 32) mraw = Mp[(size_t)b * 2048 + (it + 1) * 32 + tid];
        }

        // ---- S = Q * K^T  (two 16x16 column tiles, 4 chunks of K=32),
        //      B operands double-buffered in registers so LDS loads for
        //      chunk dc+1 overlap the WMMAs of chunk dc ----
        v8f s0 = {}, s1 = {};
        {
            const _Float16* kb0 = &sK[cur][ln * 136        + hb * 8];
            const _Float16* kb1 = &sK[cur][(16 + ln) * 136 + hb * 8];
            v16h b0c = load16h(kb0);
            v16h b1c = load16h(kb1);
#pragma unroll
            for (int dc = 0; dc < 4; ++dc) {
                v16h b0n = b0c, b1n = b1c;
                if (dc < 3) {
                    b0n = load16h(kb0 + (dc + 1) * 32);
                    b1n = load16h(kb1 + (dc + 1) * 32);
                }
                s0 = __builtin_amdgcn_wmma_f32_16x16x32_f16(false, aQ[dc], false, b0c,
                                                            (short)0, s0, false, false);
                s1 = __builtin_amdgcn_wmma_f32_16x16x32_f16(false, aQ[dc], false, b1c,
                                                            (short)0, s1, false, false);
                b0c = b0n; b1c = b1n;
            }
        }

        // ---- scale + key mask (one key column per lane), spill to LDS ----
        {
            const float keep0 = sMaskF[cur][ln];
            const float keep1 = sMaskF[cur][16 + ln];
#pragma unroll
            for (int r = 0; r < 8; ++r) {
                const int row = r + hb * 8;
                sS[w][row * 33 + ln]      = (keep0 != 0.0f) ? s0[r] * SCALE : NEGV;
                sS[w][row * 33 + 16 + ln] = (keep1 != 0.0f) ? s1[r] * SCALE : NEGV;
            }
        }
        asm volatile("s_wait_dscnt 0" ::: "memory");

        // ---- online softmax: lanes 0..15 each own one row ----
        if (lane < 16) {
            const int row = lane;
            float sv[32];
            float mloc = -3.0e38f;
#pragma unroll
            for (int j = 0; j < 32; ++j) {
                sv[j] = sS[w][row * 33 + j];
                mloc = fmaxf(mloc, sv[j]);
            }
            const float mnew  = fmaxf(m_run, mloc);
            const float alpha = __expf(m_run - mnew);
            float lsum = 0.0f;
#pragma unroll
            for (int j = 0; j < 32; ++j) {
                const float p = __expf(sv[j] - mnew);
                lsum += p;
                sP[w][row * 40 + j] = (_Float16)p;
            }
            l_run = l_run * alpha + lsum;
            m_run = mnew;
            sAlpha[w][row] = alpha;
        }
        asm volatile("s_wait_dscnt 0" ::: "memory");

        // ---- rescale O and accumulate P*V (B operands double-buffered) ----
        float av[8];
#pragma unroll
        for (int r = 0; r < 8; ++r) av[r] = sAlpha[w][r + hb * 8];

        v16h pa;
        {
            v8h lo = *(const v8h*)&sP[w][ln * 40 + hb * 8];
            v8h hi = *(const v8h*)&sP[w][ln * 40 + 16 + hb * 8];
#pragma unroll
            for (int i = 0; i < 8; ++i) { pa[i] = lo[i]; pa[i + 8] = hi[i]; }
        }
        {
            const _Float16* vb = &sVt[cur][ln * 40 + hb * 8];
            v16h bvc = load16h(vb);
#pragma unroll
            for (int dt = 0; dt < 8; ++dt) {
                v16h bvn = bvc;
                if (dt < 7) bvn = load16h(vb + (dt + 1) * 16 * 40);
#pragma unroll
                for (int r = 0; r < 8; ++r) oc[dt][r] *= av[r];
                oc[dt] = __builtin_amdgcn_wmma_f32_16x16x32_f16(false, pa, false, bvc,
                                                                (short)0, oc[dt], false, false);
                bvc = bvn;
            }
        }

        // ---- commit NEXT tile into the other LDS buffer (no reader conflict) ----
        if (more) {
            _Float16 th[16];
#pragma unroll
            for (int g = 0; g < 4; ++g) {
                th[g * 4 + 0] = (_Float16)kf[g].x; th[g * 4 + 1] = (_Float16)kf[g].y;
                th[g * 4 + 2] = (_Float16)kf[g].z; th[g * 4 + 3] = (_Float16)kf[g].w;
            }
            *(v8h*)&sK[nxt][trow * 136 + tcol]     = *(v8h*)&th[0];
            *(v8h*)&sK[nxt][trow * 136 + tcol + 8] = *(v8h*)&th[8];
#pragma unroll
            for (int g = 0; g < 4; ++g) {
                sVt[nxt][(tcol + g * 4 + 0) * 40 + trow] = (_Float16)vf[g].x;
                sVt[nxt][(tcol + g * 4 + 1) * 40 + trow] = (_Float16)vf[g].y;
                sVt[nxt][(tcol + g * 4 + 2) * 40 + trow] = (_Float16)vf[g].z;
                sVt[nxt][(tcol + g * 4 + 3) * 40 + trow] = (_Float16)vf[g].w;
            }
            if (tid < 32) sMaskF[nxt][tid] = (mraw != 0) ? 1.0f : 0.0f;
        }
        __syncthreads();   // publish next tile / retire reads of current tile
    }

    // ---- epilogue: divide by l, write out ----
    if (lane < 16) sL[w][lane] = l_run;
    asm volatile("s_wait_dscnt 0" ::: "memory");

    float linv[8];
#pragma unroll
    for (int r = 0; r < 8; ++r) {
        const float lv = sL[w][r + hb * 8];
        linv[r] = (lv > 0.0f) ? (1.0f / lv) : 0.0f;
    }
    float* op = Op + ((size_t)b * 2048 + qbase) * 128;
#pragma unroll
    for (int dt = 0; dt < 8; ++dt)
#pragma unroll
        for (int r = 0; r < 8; ++r)
            op[(size_t)(r + hb * 8) * 128 + dt * 16 + ln] = oc[dt][r] * linv[r];
}

extern "C" void kernel_launch(void* const* d_in, const int* in_sizes, int n_in,
                              void* d_out, int out_size, void* d_ws, size_t ws_size,
                              hipStream_t stream) {
    (void)in_sizes; (void)n_in; (void)out_size; (void)d_ws; (void)ws_size;
    const float* q = (const float*)d_in[0];
    const float* k = (const float*)d_in[1];
    const float* v = (const float*)d_in[2];
    const int*   m = (const int*)d_in[3];
    float* out = (float*)d_out;

    dim3 grid(16, 8);   // Q/128 tiles, B batches
    dim3 block(256);    // 8 waves (wave32)
    fa_f16_wmma_kernel<<<grid, block, 0, stream>>>(q, k, v, m, out);
}
